// vFSMNCell_88562225643919
// MI455X (gfx1250) — matrix operand order repaired and checked by reading, not compiled, not run on hardware
//
#include <hip/hip_runtime.h>

// ---------------------------------------------------------------------------
// vFSMN cell for MI455X (gfx1250, wave32, WMMA, async-to-LDS DMA).
// out = X@W1 + H@W2 + bias,  H = causal depthwise conv(X, table, width 21)
// Fused as one BF16 WMMA GEMM:  out = [X|H](16384x2048) @ [W1;W2]^T-packed + bias
// ---------------------------------------------------------------------------

typedef __bf16 bf16_t;
typedef __attribute__((ext_vector_type(8)))  __bf16 v8bf;
typedef __attribute__((ext_vector_type(16))) __bf16 v16bf;
typedef __attribute__((ext_vector_type(8)))  float  v8f;

static constexpr int Bb   = 8;
static constexpr int Nn   = 2048;
static constexpr int Dd   = 1024;
static constexpr int Mm   = Bb * Nn;     // 16384 rows
static constexpr int Kdim = 2 * Dd;      // 2048 (X | H concatenated)
static constexpr int Nout = 1024;

static constexpr int BM = 128, BN = 128, BK = 32;
static constexpr int LDA = BK + 8;       // 40 bf16 row stride in LDS (16B aligned, pad vs bank conflicts)

__device__ __forceinline__ bf16_t f2bf(float f) {
    // round-to-nearest-even fp32 -> bf16
    unsigned u = __builtin_bit_cast(unsigned, f);
    unsigned r = (u + 0x7FFFu + ((u >> 16) & 1u)) >> 16;
    return __builtin_bit_cast(bf16_t, (unsigned short)r);
}

// Flat shared-memory pointer -> raw LDS byte offset (aperture: addr[31:0] = LDS offset)
__device__ __forceinline__ unsigned lds_off(const void* p) {
    return (unsigned)(unsigned long long)(uintptr_t)p;
}

// CDNA5 async DMA: 16B per lane, global -> LDS, tracked by ASYNCcnt.
__device__ __forceinline__ void async_copy_b128(unsigned lds_byte_off, const void* gptr) {
    asm volatile("global_load_async_to_lds_b128 %0, %1, off"
                 :
                 : "v"(lds_byte_off), "v"((unsigned long long)(uintptr_t)gptr)
                 : "memory");
}

__device__ __forceinline__ void wait_asynccnt0() {
#if __has_builtin(__builtin_amdgcn_s_wait_asynccnt)
    __builtin_amdgcn_s_wait_asynccnt(0);
#else
    asm volatile("s_wait_asynccnt 0" ::: "memory");
#endif
}

// ---------------------------------------------------------------------------
// Kernel 1: depthwise causal conv (width 21, tap0 == 0) + bf16 pack of [X|H].
// Whole X (64MB) sits in the 192MB L2, so the 20 re-reads per row are cheap.
// Each thread handles 4 consecutive channels of one (b, i) row.
// ---------------------------------------------------------------------------
__global__ __launch_bounds__(256)
void conv_pack_kernel(const float* __restrict__ x,
                      const float* __restrict__ table,
                      bf16_t* __restrict__ XH)
{
    const int idx = blockIdx.x * 256 + threadIdx.x;   // over Mm * (Dd/4)
    const int m   = idx / (Dd / 4);
    const int d4  = (idx % (Dd / 4)) * 4;
    const int i   = m & (Nn - 1);                     // position within sequence

    const float4 xv = *(const float4*)(x + (size_t)m * Dd + d4);

    float4 h = make_float4(0.f, 0.f, 0.f, 0.f);
    const int kmax = i < 20 ? i : 20;
    for (int k = 1; k <= kmax; ++k) {
        const float4 tv = *(const float4*)(table + (size_t)k * Dd + d4);
        const float4 xp = *(const float4*)(x + (size_t)(m - k) * Dd + d4);
        h.x = fmaf(tv.x, xp.x, h.x);
        h.y = fmaf(tv.y, xp.y, h.y);
        h.z = fmaf(tv.z, xp.z, h.z);
        h.w = fmaf(tv.w, xp.w, h.w);
    }

    bf16_t* px = XH + (size_t)m * Kdim + d4;          // X part (k in [0,1024))
    px[0] = f2bf(xv.x); px[1] = f2bf(xv.y); px[2] = f2bf(xv.z); px[3] = f2bf(xv.w);
    bf16_t* ph = XH + (size_t)m * Kdim + Dd + d4;     // H part (k in [1024,2048))
    ph[0] = f2bf(h.x);  ph[1] = f2bf(h.y);  ph[2] = f2bf(h.z);  ph[3] = f2bf(h.w);
}

// ---------------------------------------------------------------------------
// Kernel 2: pack [W1; W2] (each D x OUT) into WT[n][k] bf16 (N-major, K contig)
// so B-matrix fragments are contiguous LDS reads.
// ---------------------------------------------------------------------------
__global__ __launch_bounds__(256)
void pack_w_kernel(const float* __restrict__ W1,
                   const float* __restrict__ W2,
                   bf16_t* __restrict__ WT)
{
    const int idx = blockIdx.x * 256 + threadIdx.x;   // over Nout * Kdim
    const int n = idx >> 11;                          // / 2048
    const int k = idx & (Kdim - 1);
    const float w = (k < Dd) ? W1[(size_t)k * Nout + n]
                             : W2[(size_t)(k - Dd) * Nout + n];
    WT[idx] = f2bf(w);
}

// ---------------------------------------------------------------------------
// Kernel 3: BF16 WMMA GEMM with async-to-LDS DMA double buffering.
//   C(16384x1024) = XH(16384x2048) * WT^T + bias
// Block: 128x128 tile, BK=32, 256 threads = 8 waves.
// Wave w: rows (w>>2)*64 .. +63, cols (w&3)*32 .. +31  -> 4x2 wmma tiles.
// Per K-step: issue DMA for tile kt+1, run 8 WMMAs on tile kt, then
// s_wait_asynccnt 0 + barrier. No VGPR staging, no loadcnt stalls in the loop.
// ---------------------------------------------------------------------------
__global__ __launch_bounds__(256)
void gemm_bf16_kernel(const bf16_t* __restrict__ XH,
                      const bf16_t* __restrict__ WT,
                      const float*  __restrict__ bias,
                      float* __restrict__ out)
{
    __shared__ bf16_t As[2][BM * LDA];
    __shared__ bf16_t Bs[2][BN * LDA];

    const int tid  = threadIdx.x;
    const int lane = tid & 31;
    const int wave = tid >> 5;
    const int wr   = wave >> 2;        // 0..1
    const int wc   = wave & 3;         // 0..3

    const int blockN = blockIdx.x * BN;
    const int blockM = blockIdx.y * BM;

    v8f zero8;
    #pragma unroll
    for (int e = 0; e < 8; ++e) zero8[e] = 0.0f;

    v8f acc[4][2];
    #pragma unroll
    for (int mi = 0; mi < 4; ++mi)
        #pragma unroll
        for (int ni = 0; ni < 2; ++ni) acc[mi][ni] = zero8;

    // ---- DMA issue: tile = 128 rows x 32 k = 512 x 16B chunks; 256 threads
    //      issue 2 chunks of A + 2 chunks of B each (ASYNCcnt-tracked).
    const int ldRow = tid >> 2;              // 0..63 (+64 on second chunk)
    const int ldSeg = (tid & 3) * 8;         // bf16 column within 32-wide tile
    auto issue_tile = [&](int k0, int buf) {
        #pragma unroll
        for (int j = 0; j < 2; ++j) {
            const int row = ldRow + 64 * j;
            async_copy_b128(lds_off(&As[buf][row * LDA + ldSeg]),
                            XH + (size_t)(blockM + row) * Kdim + k0 + ldSeg);
            async_copy_b128(lds_off(&Bs[buf][row * LDA + ldSeg]),
                            WT + (size_t)(blockN + row) * Kdim + k0 + ldSeg);
        }
    };

    issue_tile(0, 0);
    wait_asynccnt0();
    __syncthreads();

    const int fr  = lane & 15;
    const int hi  = lane >> 4;
    const int akb = hi * 16;   // A frag K base (contiguous K, matches B)
    const int bkb = hi * 16;   // B frag K base (contiguous K, matches A)

    const int KT = Kdim / BK;  // 64 K-steps
    for (int kt = 0; kt < KT; ++kt) {
        const int cur = kt & 1;
        if (kt + 1 < KT) issue_tile((kt + 1) * BK, cur ^ 1);   // DMA overlaps WMMA

        // ---- build fragments in contiguous K order (k = 16*hi + e), same for A and B
        v16bf afrag[4];
        #pragma unroll
        for (int mi = 0; mi < 4; ++mi) {
            const bf16_t* ap = &As[cur][(wr * 64 + mi * 16 + fr) * LDA];
            const v8bf alo = *(const v8bf*)(ap + akb);        // K = akb .. akb+7
            const v8bf ahi = *(const v8bf*)(ap + akb + 8);    // K = akb+8 .. akb+15
            v16bf af;
            #pragma unroll
            for (int e = 0; e < 8; ++e) { af[e] = alo[e]; af[e + 8] = ahi[e]; }
            afrag[mi] = af;
        }
        v16bf bfrag[2];
        #pragma unroll
        for (int ni = 0; ni < 2; ++ni) {
            const bf16_t* bp = &Bs[cur][(wc * 32 + ni * 16 + fr) * LDA + bkb];
            const v8bf blo = *(const v8bf*)(bp);              // K = bkb .. bkb+7
            const v8bf bhi = *(const v8bf*)(bp + 8);          // K = bkb+8 .. bkb+15
            v16bf bf_;
            #pragma unroll
            for (int e = 0; e < 8; ++e) { bf_[e] = blo[e]; bf_[e + 8] = bhi[e]; }
            bfrag[ni] = bf_;
        }

        // ---- 8 WMMAs per wave per K-step
        #pragma unroll
        for (int mi = 0; mi < 4; ++mi)
            #pragma unroll
            for (int ni = 0; ni < 2; ++ni)
                acc[mi][ni] = __builtin_amdgcn_wmma_f32_16x16x32_bf16(
                    /*neg_a=*/false, afrag[mi],
                    /*neg_b=*/false, bfrag[ni],
                    /*c_mod=*/(short)0, acc[mi][ni],
                    /*reuse_a=*/false, /*reuse_b=*/false);

        if (kt + 1 < KT) {
            wait_asynccnt0();     // this wave's DMA into buf nxt landed in LDS
            __syncthreads();      // everyone's landed; prior reads of nxt done
        }
    }

    // ---- epilogue: bias + non-temporal store (output is write-once, 64MB —
    //      don't evict the L2-resident X/H working set)
    #pragma unroll
    for (int ni = 0; ni < 2; ++ni) {
        const int col = blockN + wc * 32 + ni * 16 + fr;
        const float bv = bias[col];
        #pragma unroll
        for (int mi = 0; mi < 4; ++mi) {
            const int rbase = blockM + wr * 64 + mi * 16 + hi * 8;
            #pragma unroll
            for (int v = 0; v < 8; ++v) {
                __builtin_nontemporal_store(acc[mi][ni][v] + bv,
                                            out + (size_t)(rbase + v) * Nout + col);
            }
        }
    }
}

// ---------------------------------------------------------------------------
// Launch: inputs in setup_inputs() order:
//   0: input_data (8,2048,1024) f32   1: W1 (1024,1024) f32
//   2: W2 (1024,1024) f32             3: bias (1024,) f32
//   4: embeddings_table (21,1024) f32
// Workspace: XH bf16 (16384x2048) = 64MB, WT bf16 (1024x2048) = 4MB.
// ---------------------------------------------------------------------------
extern "C" void kernel_launch(void* const* d_in, const int* in_sizes, int n_in,
                              void* d_out, int out_size, void* d_ws, size_t ws_size,
                              hipStream_t stream) {
    const float* x     = (const float*)d_in[0];
    const float* W1    = (const float*)d_in[1];
    const float* W2    = (const float*)d_in[2];
    const float* bias  = (const float*)d_in[3];
    const float* table = (const float*)d_in[4];
    float* out = (float*)d_out;

    bf16_t* XH = (bf16_t*)d_ws;
    bf16_t* WT = (bf16_t*)((char*)d_ws + (size_t)Mm * Kdim * sizeof(bf16_t));

    conv_pack_kernel<<<Mm * (Dd / 4) / 256, 256, 0, stream>>>(x, table, XH);
    pack_w_kernel<<<Nout * Kdim / 256, 256, 0, stream>>>(W1, W2, WT);

    dim3 grid(Nout / BN, Mm / BM);   // (8, 128)
    gemm_bf16_kernel<<<grid, 256, 0, stream>>>(XH, WT, bias, out);
}
